// MixBlock_4234837754378
// MI455X (gfx1250) — matrix operand order repaired and checked by reading, not compiled, not run on hardware
//
#include <hip/hip_runtime.h>
#include <math.h>

// ---------------------------------------------------------------------------
// Types
// ---------------------------------------------------------------------------
typedef unsigned short u16;
typedef __attribute__((ext_vector_type(8)))  u16    u16x8;
typedef __attribute__((ext_vector_type(16))) u16    u16x16;
typedef __attribute__((ext_vector_type(16))) __bf16 bf16x16;
typedef __attribute__((ext_vector_type(8)))  float  f32x8;
typedef __attribute__((ext_vector_type(4)))  unsigned int u32x4;
typedef __attribute__((ext_vector_type(8)))  int    i32x8;
typedef __attribute__((ext_vector_type(4)))  int    i32x4;

#define CDIM   384
#define HDIM   56
#define HWPIX  3136          // 56*56
#define NTOK   50176         // 16*3136
#define NHEADS 12
#define HEADD  32

// ---------------------------------------------------------------------------
// Helpers
// ---------------------------------------------------------------------------
__device__ inline u16 f2bf(float f) {
  unsigned u = __float_as_uint(f);
  u += 0x7fffu + ((u >> 16) & 1u);          // round-to-nearest-even
  return (u16)(u >> 16);
}
__device__ inline float bf2f(u16 h) { return __uint_as_float(((unsigned)h) << 16); }
__device__ inline float gelu_f(float x) { return 0.5f * x * (1.0f + erff(x * 0.70710678118654752f)); }

// windowed token t -> (batch, pixel-y, pixel-x)
__device__ inline void tok_decode(int t, int& b, int& py, int& px) {
  b = t / HWPIX;
  int r = t - b * HWPIX;
  int win = r / 49;
  int pos = r - win * 49;
  int wy = win >> 3, wx = win & 7;
  int iy = pos / 7, ix = pos - iy * 7;
  py = wy * 7 + iy;
  px = wx * 7 + ix;
}
__device__ inline int tok2pix(int t) {
  int b, py, px; tok_decode(t, b, py, px);
  return b * HWPIX + py * HDIM + px;
}

// WMMA fragment loaders (bf16, 16x16x32).  ISA layouts:
//   A 16x32: lanes 0-15 row=lane, K={kk..kk+7, kk+16..kk+23};
//            lanes 16-31 row=lane-16, K={kk+8..kk+15, kk+24..kk+31}
//   B 32x16: same K layout as A, col=lane&15
__device__ inline bf16x16 pack_frag(u16x8 lo, u16x8 hi) {
  u16x16 v;
#pragma unroll
  for (int i = 0; i < 8; i++) { v[i] = lo[i]; v[8 + i] = hi[i]; }
  return __builtin_bit_cast(bf16x16, v);
}
__device__ inline bf16x16 load_a_frag(const u16* __restrict__ X, int K, int rowBase, int kk, int lane) {
  int row = rowBase + (lane & 15);
  int k0  = kk + ((lane >> 4) << 3);
  const u16* p = X + (size_t)row * K + k0;
  return pack_frag(*(const u16x8*)p, *(const u16x8*)(p + 16));
}
__device__ inline bf16x16 load_b_frag(const u16* __restrict__ W, int K, int colBase, int kk, int lane) {
  int col = colBase + (lane & 15);
  int kb  = kk + ((lane >> 4) << 3);
  const u16* p = W + (size_t)col * K + kb;
  return pack_frag(*(const u16x8*)p, *(const u16x8*)(p + 16));
}
// LDS-staged fragments (tile stored as 64 rows x 32 bf16, row stride 32)
__device__ inline bf16x16 lds_a_frag(const u16* sA, int mt, int lane) {
  int rl = mt * 16 + (lane & 15);
  int k0 = (lane >> 4) << 3;
  const u16* p = sA + rl * 32 + k0;
  return pack_frag(*(const u16x8*)p, *(const u16x8*)(p + 16));
}
__device__ inline bf16x16 lds_b_frag(const u16* sB, int colLocal, int lane) {
  int cl = colLocal + (lane & 15);
  int kb = (lane >> 4) << 3;
  const u16* p = sB + cl * 32 + kb;
  return pack_frag(*(const u16x8*)p, *(const u16x8*)(p + 16));
}
__device__ inline f32x8 wmma_bf16(bf16x16 a, bf16x16 b, f32x8 c) {
  return __builtin_amdgcn_wmma_f32_16x16x32_bf16(false, a, false, b, (short)0, c, false, false);
}

// ---------------------------------------------------------------------------
// Tensor Data Mover: DMA a 64-row x 32-element bf16 tile (row stride = K
// elements) from global memory into LDS.  D# layout per CDNA5 ISA ch.8.
// 6-arg builtin form: (g0, g1, g2, g3, g_extra, cpol); groups 2/3/extra are
// zero for a <=2-D tensor (VADDR2/VADDR3 disabled).
// ---------------------------------------------------------------------------
__device__ inline void tdm_load_tile_64x32(unsigned ldsOff, const u16* gptr, int Kelems) {
  unsigned long long ga = (unsigned long long)(uintptr_t)gptr;
  unsigned K = (unsigned)Kelems;
  u32x4 g0;
  g0[0] = 1u;                                                // count=1 (user descriptor)
  g0[1] = ldsOff;                                            // lds_addr (bytes)
  g0[2] = (unsigned)ga;                                      // global_addr[31:0]
  g0[3] = (unsigned)((ga >> 32) & 0x01ffffffu) | (2u << 30); // global_addr[56:32] | type=2
  i32x8 g1;
  g1[0] = (int)(1u << 16);                 // workgroup_mask=0, data_size=1 (2 bytes)
  g1[1] = (int)((K & 0xffffu) << 16);      // tensor_dim0[15:0]
  g1[2] = (int)((K >> 16) & 0xffffu);      // tensor_dim0[31:16] | tensor_dim1[15:0]=0
  g1[3] = (int)(16u | (32u << 16));        // tensor_dim1[31:16]=16 (dim1=1M rows) | tile_dim0=32
  g1[4] = (int)64u;                        // tile_dim1=64, tile_dim2=0
  g1[5] = (int)K;                          // tensor_dim0_stride[31:0]
  g1[6] = 0;                               // tensor_dim0_stride[47:32]=0, dim1_stride lo=0
  g1[7] = 0;
  i32x4 z4 = {0, 0, 0, 0};
  i32x8 z8 = {0, 0, 0, 0, 0, 0, 0, 0};
  __builtin_amdgcn_tensor_load_to_lds(g0, g1, z4, z4, z8, 0);
}

__device__ inline float blockReduce128(float v, float* red) {
  red[threadIdx.x] = v; __syncthreads();
#pragma unroll
  for (int off = 64; off > 0; off >>= 1) {
    if ((int)threadIdx.x < off) red[threadIdx.x] += red[threadIdx.x + off];
    __syncthreads();
  }
  float r = red[0]; __syncthreads();
  return r;
}

// ---------------------------------------------------------------------------
// Small prep kernels
// ---------------------------------------------------------------------------
__global__ __launch_bounds__(256) void cvt_kernel(const float* __restrict__ s, u16* __restrict__ d, int n) {
  int i = blockIdx.x * 256 + threadIdx.x;
  if (i < n) d[i] = f2bf(s[i]);
}
__global__ __launch_bounds__(256) void bnprep_kernel(const float* g, const float* b, const float* m,
                                                     const float* v, float* sc, float* sh, int n) {
  int i = blockIdx.x * 256 + threadIdx.x;
  if (i < n) {
    float s = g[i] * rsqrtf(v[i] + 1e-5f);
    sc[i] = s;
    sh[i] = b[i] - m[i] * s;
  }
}

// LN(n1) over each token, permuted to windowed order, bf16 output.
__global__ __launch_bounds__(128) void ln1_win_kernel(const float* __restrict__ x,
                                                      const float* __restrict__ g,
                                                      const float* __restrict__ b,
                                                      u16* __restrict__ out) {
  __shared__ float red[128];
  int t = blockIdx.x;
  const float* row = x + (size_t)tok2pix(t) * CDIM;
  float va[3];
  float s = 0.f, q = 0.f;
#pragma unroll
  for (int j = 0; j < 3; j++) {
    float v = row[threadIdx.x + j * 128];
    va[j] = v; s += v; q += v * v;
  }
  s = blockReduce128(s, red);
  q = blockReduce128(q, red);
  float mean = s * (1.f / CDIM);
  float rstd = rsqrtf(q * (1.f / CDIM) - mean * mean + 1e-5f);
#pragma unroll
  for (int j = 0; j < 3; j++) {
    int c = threadIdx.x + j * 128;
    out[(size_t)t * CDIM + c] = f2bf((va[j] - mean) * rstd * g[c] + b[c]);
  }
}

// ---------------------------------------------------------------------------
// Row-block GEMM with fused LayerNorm (N fixed = 384).
// EPI 0: LN -> bf16 row-major (windowed)
// EPI 1: LN -> f32 NCHW spatial scatter
// EPI 2: bias + shortcut residual; store x_res (f32 windowed) and LN -> bf16
// ---------------------------------------------------------------------------
template <int EPI>
__global__ __launch_bounds__(256) void gemm_ln384_kernel(
    const u16* __restrict__ X, const u16* __restrict__ W, const float* __restrict__ bias,
    const float* __restrict__ g, const float* __restrict__ beta, int K,
    u16* __restrict__ outH, float* __restrict__ outF,
    const float* __restrict__ shortcut, float* __restrict__ xres) {
  __shared__ float tile[16][CDIM];
  __shared__ float red0[256], red1[256];
  __shared__ float smean[16], srstd[16];

  int lane = threadIdx.x & 31;
  int wv   = threadIdx.x >> 5;          // 0..7
  int rowBase = blockIdx.x * 16;

  f32x8 acc[3] = {};
  for (int kk = 0; kk < K; kk += 32) {
    bf16x16 a = load_a_frag(X, K, rowBase, kk, lane);
#pragma unroll
    for (int t = 0; t < 3; t++) {
      bf16x16 b = load_b_frag(W, K, (wv + t * 8) * 16, kk, lane);
      acc[t] = wmma_bf16(a, b, acc[t]);
    }
  }
  int colL = lane & 15;
  int rOff = (lane >> 4) << 3;
#pragma unroll
  for (int t = 0; t < 3; t++) {
    int col = (wv + t * 8) * 16 + colL;
    float bv = bias[col];
#pragma unroll
    for (int r = 0; r < 8; r++) tile[rOff + r][col] = acc[t][r] + bv;
  }
  __syncthreads();

  int rid = threadIdx.x >> 4;           // 0..15 (row)
  int sub = threadIdx.x & 15;           // 0..15 (col group)

  if (EPI == 2) {
    int t = rowBase + rid;
    int b, py, px; tok_decode(t, b, py, px);
    size_t orig = ((size_t)(b * HWPIX + py * HDIM + px)) * CDIM;
#pragma unroll
    for (int i = 0; i < 24; i++) {
      int col = sub * 24 + i;
      float v = tile[rid][col] + shortcut[orig + col];
      tile[rid][col] = v;
      xres[(size_t)t * CDIM + col] = v;
    }
    __syncthreads();
  }

  float s = 0.f, q = 0.f;
#pragma unroll
  for (int i = 0; i < 24; i++) {
    float v = tile[rid][sub * 24 + i];
    s += v; q += v * v;
  }
  red0[threadIdx.x] = s; red1[threadIdx.x] = q;
  __syncthreads();
#pragma unroll
  for (int off = 8; off > 0; off >>= 1) {
    if (sub < off) {
      red0[threadIdx.x] += red0[threadIdx.x + off];
      red1[threadIdx.x] += red1[threadIdx.x + off];
    }
    __syncthreads();
  }
  if (sub == 0) {
    float m = red0[threadIdx.x] * (1.f / CDIM);
    float var = red1[threadIdx.x] * (1.f / CDIM) - m * m;
    smean[rid] = m;
    srstd[rid] = rsqrtf(var + 1e-5f);
  }
  __syncthreads();
  float mean = smean[rid], rstd = srstd[rid];
  int t = rowBase + rid;
  int bb = 0, py = 0, px = 0;
  if (EPI == 1) tok_decode(t, bb, py, px);
#pragma unroll
  for (int i = 0; i < 24; i++) {
    int col = sub * 24 + i;
    float v = (tile[rid][col] - mean) * rstd * g[col] + beta[col];
    if (EPI == 0 || EPI == 2) outH[(size_t)t * CDIM + col] = f2bf(v);
    else                      outF[((size_t)(bb * CDIM + col)) * HWPIX + py * HDIM + px] = v;
  }
}

// ---------------------------------------------------------------------------
// Generic 64x64-tile GEMM, TDM-staged double-buffered LDS pipeline.
// Wave 0 issues TENSOR_LOAD_TO_LDS DMAs for the A (64x32) and B (64x32)
// bf16 tiles of the next K-step while all 8 waves compute WMMA on the
// current LDS buffers.
// MODE 0: +bias -> bf16 row-major      MODE 1: +bias, BN affine -> f32 (ldo,colOff)
// MODE 2: +bias, GELU -> bf16          MODE 4: +bias + residual -> f32, token-order scatter
// MODE 5: +bias, BN affine, GELU -> bf16
// ---------------------------------------------------------------------------
template <int MODE>
__global__ __launch_bounds__(256) void gemm64_kernel(
    const u16* __restrict__ X, const u16* __restrict__ W, const float* __restrict__ bias,
    int M, int N, int K,
    u16* __restrict__ outH, float* __restrict__ outF, int ldo, int colOff,
    const float* __restrict__ bnS, const float* __restrict__ bnB,
    const float* __restrict__ resid) {
  __shared__ u16 sA[2][64 * 32];
  __shared__ u16 sB[2][64 * 32];

  int lane = threadIdx.x & 31;
  int wv   = threadIdx.x >> 5;
  int mt = wv & 3;
  int ng = wv >> 2;
  int rowBase = blockIdx.y * 64 + mt * 16;
  int col0    = blockIdx.x * 64 + ng * 32;
  const u16* Ablk = X + (size_t)(blockIdx.y * 64) * K;
  const u16* Bblk = W + (size_t)(blockIdx.x * 64) * K;

  int nsteps = K / 32;
  if (wv == 0) {
    tdm_load_tile_64x32((unsigned)(uintptr_t)&sA[0][0], Ablk, K);
    tdm_load_tile_64x32((unsigned)(uintptr_t)&sB[0][0], Bblk, K);
  }
  f32x8 acc0 = {}, acc1 = {};
  for (int s = 0; s < nsteps; s++) {
    int cur = s & 1;
    if (wv == 0) {
      if (s + 1 < nsteps) {
        tdm_load_tile_64x32((unsigned)(uintptr_t)&sA[cur ^ 1][0], Ablk + (s + 1) * 32, K);
        tdm_load_tile_64x32((unsigned)(uintptr_t)&sB[cur ^ 1][0], Bblk + (s + 1) * 32, K);
        __builtin_amdgcn_s_wait_tensorcnt(2);   // current pair complete (in-order)
      } else {
        __builtin_amdgcn_s_wait_tensorcnt(0);   // drain
      }
    }
    __syncthreads();
    bf16x16 a  = lds_a_frag(&sA[cur][0], mt, lane);
    bf16x16 b0 = lds_b_frag(&sB[cur][0], ng * 32,      lane);
    bf16x16 b1 = lds_b_frag(&sB[cur][0], ng * 32 + 16, lane);
    acc0 = wmma_bf16(a, b0, acc0);
    acc1 = wmma_bf16(a, b1, acc1);
    __syncthreads();   // protect buffer before next-next DMA overwrites it
  }

  int colL = lane & 15;
  int rOff = (lane >> 4) << 3;
#pragma unroll
  for (int j = 0; j < 2; j++) {
    f32x8 acc = j ? acc1 : acc0;
    int col = col0 + j * 16 + colL;
    float bv = bias[col];
#pragma unroll
    for (int r = 0; r < 8; r++) {
      int row = rowBase + rOff + r;
      float v = acc[r] + bv;
      if (MODE == 0) {
        outH[(size_t)row * ldo + col] = f2bf(v);
      } else if (MODE == 1) {
        v = v * bnS[col] + bnB[col];
        outF[(size_t)row * ldo + colOff + col] = v;
      } else if (MODE == 2) {
        outH[(size_t)row * ldo + col] = f2bf(gelu_f(v));
      } else if (MODE == 5) {
        v = gelu_f(v * bnS[col] + bnB[col]);
        outH[(size_t)row * ldo + col] = f2bf(v);
      } else if (MODE == 4) {
        v += resid[(size_t)row * CDIM + col];
        outF[(size_t)tok2pix(row) * CDIM + col] = v;
      }
    }
  }
}

// ---------------------------------------------------------------------------
// Depthwise conv (NCHW in) + BN + GELU -> bf16 NHWC (pixel-major)
// ---------------------------------------------------------------------------
template <int KS>
__global__ __launch_bounds__(256) void dwconv_kernel(
    const float* __restrict__ in, const float* __restrict__ w, const float* __restrict__ bias,
    const float* __restrict__ bnS, const float* __restrict__ bnB, u16* __restrict__ out, int total) {
  int idx = blockIdx.x * 256 + threadIdx.x;
  if (idx >= total) return;
  int c = idx % CDIM;
  int p = idx / CDIM;
  int b = p / HWPIX;
  int pp = p - b * HWPIX;
  int y = pp / HDIM, x = pp - y * HDIM;
  const float* base = in + ((size_t)(b * CDIM + c)) * HWPIX;
  float s = bias[c];
#pragma unroll
  for (int dy = 0; dy < KS; dy++) {
    int yy = y + dy - KS / 2;
    if (yy < 0 || yy >= HDIM) continue;
#pragma unroll
    for (int dx = 0; dx < KS; dx++) {
      int xx = x + dx - KS / 2;
      if (xx < 0 || xx >= HDIM) continue;
      s += base[yy * HDIM + xx] * w[c * KS * KS + dy * KS + dx];
    }
  }
  out[(size_t)p * CDIM + c] = f2bf(gelu_f(s * bnS[c] + bnB[c]));
}

// ---------------------------------------------------------------------------
// GAP over pixels (NHWC input) and channel-gate MLP
// ---------------------------------------------------------------------------
__global__ __launch_bounds__(384) void gap_kernel(const float* __restrict__ xcnn, float* __restrict__ gap) {
  int b = blockIdx.x, c = threadIdx.x;
  const float* p = xcnn + (size_t)b * HWPIX * CDIM + c;
  float s = 0.f;
  for (int i = 0; i < HWPIX; i++) s += p[(size_t)i * CDIM];
  gap[b * CDIM + c] = s * (1.f / HWPIX);
}
__global__ __launch_bounds__(384) void ci_kernel(
    const float* __restrict__ gap, const float* __restrict__ w1, const float* __restrict__ b1,
    const float* __restrict__ bnS, const float* __restrict__ bnB,
    const float* __restrict__ w2, const float* __restrict__ b2, float* __restrict__ gate) {
  __shared__ float hsh[192];
  int b = blockIdx.x, tid = threadIdx.x;
  if (tid < 192) {
    float s = b1[tid];
    const float* wr = w1 + (size_t)tid * CDIM;
    for (int k = 0; k < CDIM; k++) s += gap[b * CDIM + k] * wr[k];
    hsh[tid] = gelu_f(s * bnS[tid] + bnB[tid]);
  }
  __syncthreads();
  float o = b2[tid];
  const float* wr = w2 + (size_t)tid * 192;
  for (int k = 0; k < 192; k++) o += hsh[k] * wr[k];
  gate[b * CDIM + tid] = 1.f / (1.f + expf(-o));
}

// ---------------------------------------------------------------------------
// Window attention: one block per (window, head). 49 tokens, hd=32.
// ---------------------------------------------------------------------------
__global__ __launch_bounds__(128) void attn_kernel(
    const u16* __restrict__ qkv, const float* __restrict__ gate, const float* __restrict__ rpb,
    float* __restrict__ outF, u16* __restrict__ outH) {
  __shared__ float sq[49 * 32], sk[49 * 32], sv[49 * 32];
  __shared__ float ss[49 * 49];
  int win = blockIdx.x, h = blockIdx.y;
  int b = win >> 6;
  const float qscale = 0.17677669529663687f;   // 1/sqrt(32)
  for (int e = threadIdx.x; e < 49 * 32; e += 128) {
    int i = e >> 5, d = e & 31;
    size_t base = (size_t)(win * 49 + i) * 1152 + h * 32 + d;
    sq[e] = bf2f(qkv[base]) * qscale;
    sk[e] = bf2f(qkv[base + 384]);
    sv[e] = bf2f(qkv[base + 768]) * gate[b * CDIM + h * 32 + d];
  }
  __syncthreads();
  for (int e = threadIdx.x; e < 49 * 49; e += 128) {
    int i = e / 49, j = e - i * 49;
    float dot = 0.f;
#pragma unroll
    for (int d = 0; d < 32; d++) dot += sq[i * 32 + d] * sk[j * 32 + d];
    int qy = i / 7, qx = i - qy * 7, ky = j / 7, kx = j - ky * 7;
    int rel = (qy - ky + 6) * 13 + (qx - kx + 6);
    ss[e] = dot + rpb[rel * NHEADS + h];
  }
  __syncthreads();
  if (threadIdx.x < 49) {
    float* row = ss + threadIdx.x * 49;
    float mx = row[0];
    for (int j = 1; j < 49; j++) mx = fmaxf(mx, row[j]);
    float sum = 0.f;
    for (int j = 0; j < 49; j++) { float e = expf(row[j] - mx); row[j] = e; sum += e; }
    float inv = 1.f / sum;
    for (int j = 0; j < 49; j++) row[j] *= inv;
  }
  __syncthreads();
  for (int e = threadIdx.x; e < 49 * 32; e += 128) {
    int i = e >> 5, d = e & 31;
    float acc = 0.f;
    for (int j = 0; j < 49; j++) acc += ss[i * 49 + j] * sv[j * 32 + d];
    size_t idx = (size_t)(win * 49 + i) * CDIM + h * 32 + d;
    outF[idx] = acc;
    outH[idx] = f2bf(acc);
  }
}

// ---------------------------------------------------------------------------
// Build cat = [LN_an(x_att), LN_cn(sigmoid(si) * x_cnn)] (bf16, windowed)
// ---------------------------------------------------------------------------
__global__ __launch_bounds__(128) void catbuild_kernel(
    const float* __restrict__ xatt, const u16* __restrict__ sih,
    const float* __restrict__ si2w, const float* __restrict__ si2b,
    const float* __restrict__ xcnn,
    const float* __restrict__ ang, const float* __restrict__ anb,
    const float* __restrict__ cng, const float* __restrict__ cnb,
    u16* __restrict__ cat) {
  __shared__ float red[128];
  int t = blockIdx.x;
  float va[3], vc[3];
#pragma unroll
  for (int j = 0; j < 3; j++) va[j] = xatt[(size_t)t * CDIM + threadIdx.x + j * 128];

  float ps = 0.f;
  for (int i = threadIdx.x; i < 192; i += 128) ps += bf2f(sih[(size_t)t * 192 + i]) * si2w[i];
  ps = blockReduce128(ps, red);
  float sg = 1.f / (1.f + expf(-(ps + si2b[0])));

  int p = tok2pix(t);
#pragma unroll
  for (int j = 0; j < 3; j++) vc[j] = sg * xcnn[(size_t)p * CDIM + threadIdx.x + j * 128];

  float s = 0.f, q = 0.f;
#pragma unroll
  for (int j = 0; j < 3; j++) { s += va[j]; q += va[j] * va[j]; }
  s = blockReduce128(s, red); q = blockReduce128(q, red);
  float m1 = s * (1.f / CDIM), r1 = rsqrtf(q * (1.f / CDIM) - m1 * m1 + 1e-5f);

  s = 0.f; q = 0.f;
#pragma unroll
  for (int j = 0; j < 3; j++) { s += vc[j]; q += vc[j] * vc[j]; }
  s = blockReduce128(s, red); q = blockReduce128(q, red);
  float m2 = s * (1.f / CDIM), r2 = rsqrtf(q * (1.f / CDIM) - m2 * m2 + 1e-5f);

#pragma unroll
  for (int j = 0; j < 3; j++) {
    int c = threadIdx.x + j * 128;
    cat[(size_t)t * 768 + c]       = f2bf((va[j] - m1) * r1 * ang[c] + anb[c]);
    cat[(size_t)t * 768 + 384 + c] = f2bf((vc[j] - m2) * r2 * cng[c] + cnb[c]);
  }
}

// ---------------------------------------------------------------------------
// Host launcher
// ---------------------------------------------------------------------------
extern "C" void kernel_launch(void* const* d_in, const int* in_sizes, int n_in,
                              void* d_out, int out_size, void* d_ws, size_t ws_size,
                              hipStream_t stream) {
  (void)in_sizes; (void)n_in; (void)out_size; (void)ws_size;
  auto fin = [&](int i) { return (const float*)d_in[i]; };

  const float* x_in  = fin(0);
  const float* pa_w = fin(1);  const float* pa_b = fin(2);  const float* pan_g = fin(3);  const float* pan_b = fin(4);
  const float* pc1_w = fin(5); const float* pc1_b = fin(6); const float* pc1n_g = fin(7); const float* pc1n_b = fin(8);
  const float* pc2_w = fin(9); const float* pc2_b = fin(10);const float* pc2n_g = fin(11);const float* pc2n_b = fin(12);
  const float* dw3_w = fin(13);const float* dw3_b = fin(14);
  const float* bn3_g = fin(15);const float* bn3_b = fin(16);const float* bn3_m = fin(17);const float* bn3_v = fin(18);
  const float* dw5_w = fin(19);const float* dw5_b = fin(20);
  const float* bn5_g = fin(21);const float* bn5_b = fin(22);const float* bn5_m = fin(23);const float* bn5_v = fin(24);
  const float* ci1_w = fin(25);const float* ci1_b = fin(26);
  const float* cibn_g= fin(27);const float* cibn_b= fin(28);const float* cibn_m= fin(29);const float* cibn_v= fin(30);
  const float* ci2_w = fin(31);const float* ci2_b = fin(32);
  const float* p1_w  = fin(33);const float* p1_b  = fin(34);
  const float* p2_w  = fin(35);const float* p2_b  = fin(36);
  const float* cn1_g = fin(37);const float* cn1_b = fin(38);const float* cn1_m = fin(39);const float* cn1_v = fin(40);
  const float* cn2_g = fin(41);const float* cn2_b = fin(42);const float* cn2_m = fin(43);const float* cn2_v = fin(44);
  const float* qkv_w = fin(45);const float* qkv_b = fin(46);
  const float* si1_w = fin(47);const float* si1_b = fin(48);
  const float* sibn_g= fin(49);const float* sibn_b= fin(50);const float* sibn_m= fin(51);const float* sibn_v= fin(52);
  const float* si2_w = fin(53);const float* si2_b = fin(54);
  const float* an_g  = fin(55);const float* an_b  = fin(56);
  const float* cn_g  = fin(57);const float* cn_b  = fin(58);
  const float* pcat_w= fin(59);const float* pcat_b= fin(60);
  const float* proj_w= fin(61);const float* proj_b= fin(62);
  const float* rpb   = fin(63);
  const float* n1_g  = fin(64);const float* n1_b  = fin(65);
  const float* n2_g  = fin(66);const float* n2_b  = fin(67);
  const float* fc1_w = fin(68);const float* fc1_b = fin(69);
  const float* fc2_w = fin(70);const float* fc2_b = fin(71);

  // --- workspace bump allocator ---
  char* wsb = (char*)d_ws;
  size_t off = 0;
  auto alloc = [&](size_t bytes) -> void* {
    void* p = wsb + off;
    off += (bytes + 255) & ~(size_t)255;
    return p;
  };
  const size_t M = NTOK;
  u16* wb_pa   = (u16*)alloc(384 * 384 * 2);
  u16* wb_pc1  = (u16*)alloc(384 * 384 * 2);
  u16* wb_pc2  = (u16*)alloc(384 * 384 * 2);
  u16* wb_qkv  = (u16*)alloc(1152 * 384 * 2);
  u16* wb_p1   = (u16*)alloc(192 * 384 * 2);
  u16* wb_p2   = (u16*)alloc(192 * 384 * 2);
  u16* wb_si1  = (u16*)alloc(192 * 384 * 2);
  u16* wb_pcat = (u16*)alloc(384 * 768 * 2);
  u16* wb_proj = (u16*)alloc(384 * 384 * 2);
  u16* wb_fc1  = (u16*)alloc(1536 * 384 * 2);
  u16* wb_fc2  = (u16*)alloc(384 * 1536 * 2);
  float* s_bn3 = (float*)alloc(384 * 4);  float* h_bn3 = (float*)alloc(384 * 4);
  float* s_bn5 = (float*)alloc(384 * 4);  float* h_bn5 = (float*)alloc(384 * 4);
  float* s_cn1 = (float*)alloc(192 * 4);  float* h_cn1 = (float*)alloc(192 * 4);
  float* s_cn2 = (float*)alloc(192 * 4);  float* h_cn2 = (float*)alloc(192 * 4);
  float* s_cib = (float*)alloc(192 * 4);  float* h_cib = (float*)alloc(192 * 4);
  float* s_sib = (float*)alloc(192 * 4);  float* h_sib = (float*)alloc(192 * 4);

  u16*   xw_bf    = (u16*)alloc(M * 384 * 2);
  u16*   xatt_bf  = (u16*)alloc(M * 384 * 2);      // LN(pa) output
  float* xc1_sp   = (float*)alloc(M * 384 * 4);    // NCHW
  float* xc2_sp   = (float*)alloc(M * 384 * 4);    // NCHW
  u16*   qkv_bf   = (u16*)alloc(M * 1152 * 2);
  u16*   xc1g_bf  = (u16*)alloc(M * 384 * 2);      // NHWC
  u16*   xc2g_bf  = (u16*)alloc(M * 384 * 2);      // NHWC
  float* xcnn     = (float*)alloc(M * 384 * 4);    // NHWC
  float* gapb     = (float*)alloc(16 * 384 * 4);
  float* gate     = (float*)alloc(16 * 384 * 4);
  float* xatto_f  = (float*)alloc(M * 384 * 4);    // attention out f32 windowed
  u16*   xatto_bf = (u16*)alloc(M * 384 * 2);
  u16*   sih_bf   = (u16*)alloc(M * 192 * 2);
  u16*   cat_bf   = (u16*)alloc(M * 768 * 2);
  u16*   pco_bf   = (u16*)alloc(M * 384 * 2);      // pcat output
  float* xres     = (float*)alloc(M * 384 * 4);    // residual (windowed)
  u16*   ln2_bf   = (u16*)alloc(M * 384 * 2);
  u16*   h_bf     = (u16*)alloc(M * 1536 * 2);

  auto cvt = [&](const float* s, u16* d, int n) {
    cvt_kernel<<<(n + 255) / 256, 256, 0, stream>>>(s, d, n);
  };
  cvt(pa_w,  wb_pa,  384 * 384);   cvt(pc1_w, wb_pc1, 384 * 384);  cvt(pc2_w, wb_pc2, 384 * 384);
  cvt(qkv_w, wb_qkv, 1152 * 384);  cvt(p1_w, wb_p1, 192 * 384);    cvt(p2_w, wb_p2, 192 * 384);
  cvt(si1_w, wb_si1, 192 * 384);   cvt(pcat_w, wb_pcat, 384 * 768);cvt(proj_w, wb_proj, 384 * 384);
  cvt(fc1_w, wb_fc1, 1536 * 384);  cvt(fc2_w, wb_fc2, 384 * 1536);

  bnprep_kernel<<<2, 256, 0, stream>>>(bn3_g, bn3_b, bn3_m, bn3_v, s_bn3, h_bn3, 384);
  bnprep_kernel<<<2, 256, 0, stream>>>(bn5_g, bn5_b, bn5_m, bn5_v, s_bn5, h_bn5, 384);
  bnprep_kernel<<<1, 256, 0, stream>>>(cn1_g, cn1_b, cn1_m, cn1_v, s_cn1, h_cn1, 192);
  bnprep_kernel<<<1, 256, 0, stream>>>(cn2_g, cn2_b, cn2_m, cn2_v, s_cn2, h_cn2, 192);
  bnprep_kernel<<<1, 256, 0, stream>>>(cibn_g, cibn_b, cibn_m, cibn_v, s_cib, h_cib, 192);
  bnprep_kernel<<<1, 256, 0, stream>>>(sibn_g, sibn_b, sibn_m, sibn_v, s_sib, h_sib, 192);

  // 1) LN(n1) + window permute
  ln1_win_kernel<<<NTOK, 128, 0, stream>>>(x_in, n1_g, n1_b, xw_bf);

  // 2) pa/pc1/pc2 GEMMs with fused LN
  gemm_ln384_kernel<0><<<NTOK / 16, 256, 0, stream>>>(xw_bf, wb_pa, pa_b, pan_g, pan_b, 384,
                                                      xatt_bf, nullptr, nullptr, nullptr);
  gemm_ln384_kernel<1><<<NTOK / 16, 256, 0, stream>>>(xw_bf, wb_pc1, pc1_b, pc1n_g, pc1n_b, 384,
                                                      nullptr, xc1_sp, nullptr, nullptr);
  gemm_ln384_kernel<1><<<NTOK / 16, 256, 0, stream>>>(xw_bf, wb_pc2, pc2_b, pc2n_g, pc2n_b, 384,
                                                      nullptr, xc2_sp, nullptr, nullptr);

  // 3) QKV projection (TDM-staged GEMM)
  gemm64_kernel<0><<<dim3(1152 / 64, NTOK / 64), 256, 0, stream>>>(
      xatt_bf, wb_qkv, qkv_b, NTOK, 1152, 384, qkv_bf, nullptr, 1152, 0, nullptr, nullptr, nullptr);

  // 4) depthwise convs + BN + GELU
  int total = NTOK * CDIM;
  dwconv_kernel<3><<<(total + 255) / 256, 256, 0, stream>>>(xc1_sp, dw3_w, dw3_b, s_bn3, h_bn3, xc1g_bf, total);
  dwconv_kernel<5><<<(total + 255) / 256, 256, 0, stream>>>(xc2_sp, dw5_w, dw5_b, s_bn5, h_bn5, xc2g_bf, total);

  // 5) 1x1 convs p1/p2 (+BN) -> x_cnn NHWC halves
  gemm64_kernel<1><<<dim3(192 / 64, NTOK / 64), 256, 0, stream>>>(
      xc1g_bf, wb_p1, p1_b, NTOK, 192, 384, nullptr, xcnn, 384, 0, s_cn1, h_cn1, nullptr);
  gemm64_kernel<1><<<dim3(192 / 64, NTOK / 64), 256, 0, stream>>>(
      xc2g_bf, wb_p2, p2_b, NTOK, 192, 384, nullptr, xcnn, 384, 192, s_cn2, h_cn2, nullptr);

  // 6) channel gate
  gap_kernel<<<16, 384, 0, stream>>>(xcnn, gapb);
  ci_kernel<<<16, 384, 0, stream>>>(gapb, ci1_w, ci1_b, s_cib, h_cib, ci2_w, ci2_b, gate);

  // 7) window attention
  attn_kernel<<<dim3(1024, NHEADS), 128, 0, stream>>>(qkv_bf, gate, rpb, xatto_f, xatto_bf);

  // 8) spatial gate hidden (si1 + BN + GELU)
  gemm64_kernel<5><<<dim3(192 / 64, NTOK / 64), 256, 0, stream>>>(
      xatto_bf, wb_si1, si1_b, NTOK, 192, 384, sih_bf, nullptr, 192, 0, s_sib, h_sib, nullptr);

  // 9) build concat [LN_an(x_att), LN_cn(sigmoid(si)*x_cnn)]
  catbuild_kernel<<<NTOK, 128, 0, stream>>>(xatto_f, sih_bf, si2_w, si2_b, xcnn,
                                            an_g, an_b, cn_g, cn_b, cat_bf);

  // 10) pcat (768 -> 384)
  gemm64_kernel<0><<<dim3(384 / 64, NTOK / 64), 256, 0, stream>>>(
      cat_bf, wb_pcat, pcat_b, NTOK, 384, 768, pco_bf, nullptr, 384, 0, nullptr, nullptr, nullptr);

  // 11) proj + residual(shortcut) + LN(n2)
  gemm_ln384_kernel<2><<<NTOK / 16, 256, 0, stream>>>(pco_bf, wb_proj, proj_b, n2_g, n2_b, 384,
                                                      ln2_bf, nullptr, x_in, xres);

  // 12) MLP
  gemm64_kernel<2><<<dim3(1536 / 64, NTOK / 64), 256, 0, stream>>>(
      ln2_bf, wb_fc1, fc1_b, NTOK, 1536, 384, h_bf, nullptr, 1536, 0, nullptr, nullptr, nullptr);
  gemm64_kernel<4><<<dim3(384 / 64, NTOK / 64), 256, 0, stream>>>(
      h_bf, wb_fc2, fc2_b, NTOK, 384, 1536, nullptr, (float*)d_out, 384, 0, nullptr, nullptr, xres);
}
